// MSDRCell_90821378441312
// MI455X (gfx1250) — matrix operand order repaired and checked by reading, not compiled, not run on hardware
//
#include <hip/hip_runtime.h>

typedef __attribute__((ext_vector_type(16))) __bf16 v16bf;
typedef __attribute__((ext_vector_type(8)))  __bf16 v8bf;
typedef __attribute__((ext_vector_type(4)))  __bf16 v4bf;
typedef __attribute__((ext_vector_type(8)))  float  v8f;

__device__ __forceinline__ v8f wmma_bf16(v16bf a, v16bf b, v8f c) {
    return __builtin_amdgcn_wmma_f32_16x16x32_bf16(false, a, false, b, (short)0, c, false, false);
}

// ---------------------------------------------------------------------------
// WMMA GEMM: C[1024 x NN] = A[1024 x K] * B[K x NN], bf16 hi/lo split (3 wmma)
// Wave computes 32x64 output (2 M-tiles x 4 N-tiles).
// MODE 0 (A packed): C = T                         (cheb first hop)
// MODE 1 (A packed): C = 2T - prev                 (cheb recurrence)
// MODE 2 (A f32, 4 m-planes): C = leaky(T + gc_b)  (gco)
// MODE 3 (A f32): C = T + bias + att ; also -> C2  (final output)
// B always packed K-major bf16 hi/lo: Bt[col][k].
// ---------------------------------------------------------------------------
template<int MODE>
__global__ void __launch_bounds__(128)
gemm_wmma(const float* __restrict__ Af,
          const __bf16* __restrict__ Ahi, const __bf16* __restrict__ Alo,
          const __bf16* __restrict__ Bhi, const __bf16* __restrict__ Blo,
          float* __restrict__ C, const float* __restrict__ P1,
          const float* __restrict__ P2, float* __restrict__ C2,
          int NN, int K,
          long sAb, long sBb, long sCb, long sP1b, long sP2b, long sC2b,
          int lda, int ldk, int ldc, long planeA, long planeB)
{
    const bool APACK = (MODE <= 1);
    const int lane = threadIdx.x & 31;
    const int wid  = blockIdx.x * (blockDim.x >> 5) + (threadIdx.x >> 5);
    const int strips = NN >> 6;          // 64-wide col strips
    const int wpb    = 32 * strips;      // waves per batch (32 rows per wave)
    if (wid >= 16 * wpb) return;
    const int b  = wid / wpb;
    const int r  = wid % wpb;
    const int tm = r / strips;           // 0..31
    const int tn = r % strips;
    const int half = lane >> 4;
    const int mrow = lane & 15;
    const int ncol = lane & 15;
    const int hb   = half * 16;

    v8f acc[2][4];
    #pragma unroll
    for (int mt = 0; mt < 2; ++mt)
        #pragma unroll
        for (int nt = 0; nt < 4; ++nt)
            #pragma unroll
            for (int i = 0; i < 8; ++i) acc[mt][nt][i] = 0.0f;

    const int ktiles = K >> 5;
    for (int kt = 0; kt < ktiles; ++kt) {
        const int m      = (MODE == 2) ? (kt >> 3) : 0;       // 8 k-tiles per m-plane
        const int klocal = (MODE == 2) ? ((kt & 7) * 32) : (kt * 32);

        // ---- A fragments (2 M-tiles) ----
        v16bf ahi[2], alo[2];
        if (APACK) {
            #pragma unroll
            for (int mt = 0; mt < 2; ++mt) {
                const long ao = (long)b * sAb + (long)(tm * 32 + mt * 16 + mrow) * lda + klocal + hb;
                v8bf h0 = *(const v8bf*)(Ahi + ao);
                v8bf h1 = *(const v8bf*)(Ahi + ao + 8);
                v8bf l0 = *(const v8bf*)(Alo + ao);
                v8bf l1 = *(const v8bf*)(Alo + ao + 8);
                ahi[mt] = __builtin_shufflevector(h0, h1, 0,1,2,3,4,5,6,7,8,9,10,11,12,13,14,15);
                alo[mt] = __builtin_shufflevector(l0, l1, 0,1,2,3,4,5,6,7,8,9,10,11,12,13,14,15);
            }
        } else {
            #pragma unroll
            for (int mt = 0; mt < 2; ++mt) {
                const float* arow = Af + (long)b * sAb + (long)m * planeA
                                  + (long)(tm * 32 + mt * 16 + mrow) * lda + klocal + hb;
                float af[16];
                *(float4*)(af + 0)  = *(const float4*)(arow + 0);
                *(float4*)(af + 4)  = *(const float4*)(arow + 4);
                *(float4*)(af + 8)  = *(const float4*)(arow + 8);
                *(float4*)(af + 12) = *(const float4*)(arow + 12);
                #pragma unroll
                for (int i = 0; i < 16; ++i) {
                    __bf16 h = (__bf16)af[i];
                    ahi[mt][i] = h;
                    alo[mt][i] = (__bf16)(af[i] - (float)h);
                }
            }
        }

        // ---- B fragments (4 N-tiles), packed K-major: 32B contiguous per lane ----
        #pragma unroll
        for (int nt = 0; nt < 4; ++nt) {
            const long bo = (long)b * sBb + (long)m * planeB
                          + (long)(tn * 64 + nt * 16 + ncol) * ldk + klocal + half * 16;
            v16bf bh = *(const v16bf*)(Bhi + bo);
            v16bf bl = *(const v16bf*)(Blo + bo);
            #pragma unroll
            for (int mt = 0; mt < 2; ++mt) {
                acc[mt][nt] = wmma_bf16(ahi[mt], bh, acc[mt][nt]);
                acc[mt][nt] = wmma_bf16(ahi[mt], bl, acc[mt][nt]);
                acc[mt][nt] = wmma_bf16(alo[mt], bh, acc[mt][nt]);
            }
        }
    }

    float* Cb = C + (long)b * sCb;
    #pragma unroll
    for (int mt = 0; mt < 2; ++mt) {
        #pragma unroll
        for (int nt = 0; nt < 4; ++nt) {
            #pragma unroll
            for (int rr = 0; rr < 8; ++rr) {
                const int row = tm * 32 + mt * 16 + half * 8 + rr;
                const int col = tn * 64 + nt * 16 + ncol;
                const long idx = (long)row * ldc + col;
                float t = acc[mt][nt][rr];
                if (MODE == 0) {
                    Cb[idx] = t;
                } else if (MODE == 1) {
                    Cb[idx] = 2.0f * t - (P1 + (long)b * sP1b)[idx];
                } else if (MODE == 2) {
                    float v = t + P1[col];
                    Cb[idx] = v >= 0.0f ? v : 0.01f * v;
                } else {
                    float v = t + P1[idx] + (P2 + (long)b * sP2b)[idx];
                    Cb[idx] = v;
                    (C2 + (long)b * sC2b)[idx] = v;
                }
            }
        }
    }
}

// ---------------------------------------------------------------------------
// Row-major f32 -> bf16 hi/lo pack (for supports): straight copy-convert.
// ---------------------------------------------------------------------------
__global__ void pack_rm_kernel(const float* __restrict__ A,
                               __bf16* __restrict__ hi, __bf16* __restrict__ lo, long n4)
{
    long i = (long)blockIdx.x * blockDim.x + threadIdx.x;
    if (i >= n4) return;
    float4 v = ((const float4*)A)[i];
    float f[4] = {v.x, v.y, v.z, v.w};
    v4bf h, l;
    #pragma unroll
    for (int j = 0; j < 4; ++j) {
        __bf16 hh = (__bf16)f[j];
        h[j] = hh;
        l[j] = (__bf16)(f[j] - (float)hh);
    }
    ((v4bf*)hi)[i] = h;
    ((v4bf*)lo)[i] = l;
}

// ---------------------------------------------------------------------------
// X[b][1024 k][256 n] f32  ->  Xt[b][256 n][1024 k] bf16 hi/lo (LDS-tiled)
// ---------------------------------------------------------------------------
__global__ void pack_transpose_kernel(const float* __restrict__ X,
                                      __bf16* __restrict__ Thi, __bf16* __restrict__ Tlo)
{
    __shared__ float tile[32][33];
    const int blk = blockIdx.x;          // 16 * 32 * 8
    const int b  = blk >> 8;
    const int rm = blk & 255;
    const int kt = rm >> 3;              // 0..31 (k tile)
    const int nt = rm & 7;               // 0..7  (n tile)
    const int t  = threadIdx.x;          // 256
    const int c  = t & 31;
    const int r0 = t >> 5;               // 0..7
    const float* Xb = X + ((long)b * 1024 + kt * 32) * 256 + nt * 32;
    #pragma unroll
    for (int j = 0; j < 4; ++j) {
        int rr = r0 + j * 8;
        tile[rr][c] = Xb[(long)rr * 256 + c];
    }
    __syncthreads();
    const long obase = ((long)b * 256 + nt * 32) * 1024 + kt * 32;
    #pragma unroll
    for (int j = 0; j < 4; ++j) {
        int rn = r0 + j * 8;             // local n
        float v = tile[c][rn];           // transposed read (padded, conflict-free)
        __bf16 h = (__bf16)v;
        Thi[obase + (long)rn * 1024 + c] = h;
        Tlo[obase + (long)rn * 1024 + c] = (__bf16)(v - (float)h);
    }
}

// ---------------------------------------------------------------------------
// Prep: conv_in (x0), cos/sin of phase, packed-transposed Gm and W (hi/lo)
// ---------------------------------------------------------------------------
__global__ void prep_kernel(const float* __restrict__ inputs,
                            const float* __restrict__ H,
                            const float* __restrict__ R,
                            const float* __restrict__ gcw,
                            const float* __restrict__ W,
                            float* __restrict__ XS,
                            float* __restrict__ COSR, float* __restrict__ SINR,
                            __bf16* __restrict__ Gthi, __bf16* __restrict__ Gtlo,
                            __bf16* __restrict__ Wthi, __bf16* __restrict__ Wtlo)
{
    const long N0 = 16L * 1024 * 256;
    const long N1 = 4L * 1024 * 64;
    const long N2 = 4L * 128 * 256;   // Gt [m][dd][c]
    const long N3 = 128L * 128;       // Wt [e][d]
    long id = (long)blockIdx.x * blockDim.x + threadIdx.x;
    if (id < N0) {
        long b = id >> 18;
        long rem = id & 262143;
        long n = rem >> 8;
        long c = rem & 255;
        const long inb = (b * 1024 + n) * 128;
        const long hbb = ((b * 4 + 3) * 1024 + n) * 128;
        float v;
        if (c < 64)       v = inputs[inb + c];
        else if (c < 128) v = H[hbb + (c - 64)];
        else if (c < 192) v = inputs[inb + 64 + (c - 128)];
        else              v = H[hbb + 64 + (c - 192)];
        XS[id] = v;
    } else if (id < N0 + N1) {
        long j = id - N0;
        float ph = R[j] * (3.14159265358979323846f * 64.0f / 14.0f);
        COSR[j] = cosf(ph);
        SINR[j] = sinf(ph);
    } else if (id < N0 + N1 + N2) {
        long j = id - N0 - N1;
        long m = j >> 15;
        long rem = j & 32767;
        long dd = rem >> 8;
        long c  = rem & 255;
        float v = gcw[dd * 1024 + c * 4 + m];
        __bf16 h = (__bf16)v;
        Gthi[j] = h;
        Gtlo[j] = (__bf16)(v - (float)h);
    } else if (id < N0 + N1 + N2 + N3) {
        long j = id - N0 - N1 - N2;
        long e = j >> 7;
        long d = j & 127;
        float v = W[d * 128 + e];
        __bf16 h = (__bf16)v;
        Wthi[j] = h;
        Wtlo[j] = (__bf16)(v - (float)h);
    }
}

// ---------------------------------------------------------------------------
// sa[b,k] = sum_{n,d} rot(H)[b,k,n,d]*att_w[n*128+d] + att_b
// ---------------------------------------------------------------------------
__global__ void att_score_kernel(const float* __restrict__ H,
                                 const float* __restrict__ COSR,
                                 const float* __restrict__ SINR,
                                 const float* __restrict__ attw,
                                 const float* __restrict__ attb,
                                 float* __restrict__ SA)
{
    const int tid = threadIdx.x;
    const int b = blockIdx.x >> 2;
    const int k = blockIdx.x & 3;
    float sum = 0.0f;
    for (int idx = tid; idx < 1024 * 64; idx += 256) {
        int n = idx >> 6, h = idx & 63;
        long hbb = ((long)(b * 4 + k) * 1024 + n) * 128;
        float reh = H[hbb + h], imh = H[hbb + 64 + h];
        long tb = ((long)k * 1024 + n) * 64 + h;
        float c = COSR[tb], s = SINR[tb];
        sum += (c * reh - s * imh) * attw[n * 128 + h]
             + (s * reh + c * imh) * attw[n * 128 + 64 + h];
    }
    __shared__ float red[256];
    red[tid] = sum;
    __syncthreads();
    for (int s2 = 128; s2 > 0; s2 >>= 1) {
        if (tid < s2) red[tid] += red[tid + s2];
        __syncthreads();
    }
    if (tid == 0) SA[blockIdx.x] = red[0] + attb[0];
}

// att[b,n,d] = sum_k softmax_k(sa)[b,k] * rot(H)[b,k,n,d]
__global__ void att_combine_kernel(const float* __restrict__ H,
                                   const float* __restrict__ COSR,
                                   const float* __restrict__ SINR,
                                   const float* __restrict__ SA,
                                   float* __restrict__ ATT)
{
    long id = (long)blockIdx.x * blockDim.x + threadIdx.x;
    if (id >= 16L * 1024 * 128) return;
    int b = (int)(id >> 17);
    int rem = (int)(id & 131071);
    int n = rem >> 7;
    int d = rem & 127;
    int h = d & 63;
    bool isIm = (d & 64) != 0;
    float s0 = SA[b * 4 + 0], s1 = SA[b * 4 + 1], s2 = SA[b * 4 + 2], s3 = SA[b * 4 + 3];
    float mx = fmaxf(fmaxf(s0, s1), fmaxf(s2, s3));
    float w[4];
    w[0] = expf(s0 - mx); w[1] = expf(s1 - mx); w[2] = expf(s2 - mx); w[3] = expf(s3 - mx);
    float den = w[0] + w[1] + w[2] + w[3];
    float a = 0.0f;
    #pragma unroll
    for (int k = 0; k < 4; ++k) {
        long hbb = ((long)(b * 4 + k) * 1024 + n) * 128;
        float reh = H[hbb + h], imh = H[hbb + 64 + h];
        long tb = ((long)k * 1024 + n) * 64 + h;
        float c = COSR[tb], s = SINR[tb];
        a += w[k] * (isIm ? (s * reh + c * imh) : (c * reh - s * imh));
    }
    ATT[id] = a / den;
}

// new_hidden[:,0:3] = hidden_states[:,1:4]
__global__ void copy_hidden_kernel(const float* __restrict__ H, float* __restrict__ NH)
{
    long id = (long)blockIdx.x * blockDim.x + threadIdx.x;
    if (id >= 16L * 3 * 1024 * 128) return;
    long b = id / 393216;
    long rem = id - b * 393216;
    NH[b * 524288 + rem] = H[b * 524288 + 131072 + rem];
}

extern "C" void kernel_launch(void* const* d_in, const int* in_sizes, int n_in,
                              void* d_out, int out_size, void* d_ws, size_t ws_size,
                              hipStream_t stream)
{
    (void)in_sizes; (void)n_in; (void)out_size; (void)ws_size;
    const float* inputs = (const float*)d_in[0];
    const float* S      = (const float*)d_in[1];
    const float* H      = (const float*)d_in[2];
    const float* W      = (const float*)d_in[3];
    const float* bvec   = (const float*)d_in[4];
    const float* R      = (const float*)d_in[5];
    const float* gcw    = (const float*)d_in[6];
    const float* gcb    = (const float*)d_in[7];
    const float* attw   = (const float*)d_in[10];
    const float* attb   = (const float*)d_in[11];
    float* out = (float*)d_out;

    // ---- workspace layout ----
    float* ws = (float*)d_ws;
    const long PLANE = 16L * 1024 * 256;
    float* XS   = ws;                          // 4 planes f32
    float* CONV = ws + 4 * PLANE;
    float* ATT  = CONV + 16L * 1024 * 128;
    float* COSR = ATT  + 16L * 1024 * 128;
    float* SINR = COSR + 4L * 1024 * 64;
    float* SA   = SINR + 4L * 1024 * 64;       // 64
    __bf16* bfbase = (__bf16*)(SA + 64);
    const long SE = 16L * 1024 * 1024;         // supports elems
    const long TE = 16L * 256 * 1024;          // transposed X elems
    __bf16* Shi  = bfbase;
    __bf16* Slo  = Shi  + SE;
    __bf16* Xthi = Slo  + SE;
    __bf16* Xtlo = Xthi + TE;
    __bf16* Gthi = Xtlo + TE;                  // 4*128*256
    __bf16* Gtlo = Gthi + 4L * 128 * 256;
    __bf16* Wthi = Gtlo + 4L * 128 * 256;      // 128*128
    __bf16* Wtlo = Wthi + 128L * 128;

    {
        long total = 16L*1024*256 + 4L*1024*64 + 4L*128*256 + 128L*128;
        int blocks = (int)((total + 255) / 256);
        prep_kernel<<<blocks, 256, 0, stream>>>(inputs, H, R, gcw, W,
                                                XS, COSR, SINR, Gthi, Gtlo, Wthi, Wtlo);
    }
    pack_rm_kernel<<<16384, 256, 0, stream>>>(S, Shi, Slo, SE / 4);

    // Chebyshev hops: x1 = S*x0 ; x2 = 2*S*x1 - x0 ; x3 = 2*S*x2 - x1
    pack_transpose_kernel<<<4096, 256, 0, stream>>>(XS, Xthi, Xtlo);
    gemm_wmma<0><<<512, 128, 0, stream>>>(nullptr, Shi, Slo, Xthi, Xtlo,
        XS + PLANE, nullptr, nullptr, nullptr,
        256, 1024, 1048576L, 262144L, 262144L, 0L, 0L, 0L, 1024, 1024, 256, 0L, 0L);

    pack_transpose_kernel<<<4096, 256, 0, stream>>>(XS + PLANE, Xthi, Xtlo);
    gemm_wmma<1><<<512, 128, 0, stream>>>(nullptr, Shi, Slo, Xthi, Xtlo,
        XS + 2 * PLANE, XS, nullptr, nullptr,
        256, 1024, 1048576L, 262144L, 262144L, 262144L, 0L, 0L, 1024, 1024, 256, 0L, 0L);

    pack_transpose_kernel<<<4096, 256, 0, stream>>>(XS + 2 * PLANE, Xthi, Xtlo);
    gemm_wmma<1><<<512, 128, 0, stream>>>(nullptr, Shi, Slo, Xthi, Xtlo,
        XS + 3 * PLANE, XS + PLANE, nullptr, nullptr,
        256, 1024, 1048576L, 262144L, 262144L, 262144L, 0L, 0L, 1024, 1024, 256, 0L, 0L);

    // rotation attention (independent)
    att_score_kernel<<<64, 256, 0, stream>>>(H, COSR, SINR, attw, attb, SA);
    att_combine_kernel<<<8192, 256, 0, stream>>>(H, COSR, SINR, SA, ATT);

    // conv_out = leaky(sum_m xs[m] @ Gm + gc_b)   (A: f32 m-planes; B: packed Gt)
    gemm_wmma<2><<<256, 128, 0, stream>>>(XS, nullptr, nullptr, Gthi, Gtlo,
        CONV, gcb, nullptr, nullptr,
        128, 1024, 262144L, 0L, 131072L, 0L, 0L, 0L, 256, 256, 128, PLANE, 32768L);

    // output = conv_out @ W + b + att  (also -> new_hidden[:,3])
    gemm_wmma<3><<<256, 128, 0, stream>>>(CONV, nullptr, nullptr, Wthi, Wtlo,
        out, bvec, ATT, out + 2097152L + 393216L,
        128, 128, 131072L, 0L, 131072L, 0L, 131072L, 524288L, 128, 128, 128, 0L, 0L);

    copy_hidden_kernel<<<24576, 256, 0, stream>>>(H, out + 2097152L);
}
